// Model_13520557047963
// MI455X (gfx1250) — matrix operand order repaired and checked by reading, not compiled, not run on hardware
//
#include <hip/hip_runtime.h>
#include <hip/hip_bf16.h>
#include <math.h>

typedef __attribute__((ext_vector_type(16))) _Float16 v16h;
typedef __attribute__((ext_vector_type(8)))  float    v8f;

#define WMMA_F32_F16(a,b,c) \
  __builtin_amdgcn_wmma_f32_16x16x32_f16(false,(a),false,(b),(short)0,(c),false,false)

constexpr int BBATCH = 8;
constexpr int LSEQ   = 1024;
constexpr int DMODEL = 512;
constexpr int DINNER = 1024;
constexpr int DSTATE = 16;
constexpr int DTRANK = 32;
constexpr int NHEADS = 8;
constexpr int DHEAD  = 64;
constexpr int MROWS  = BBATCH * LSEQ;   // 8192

// ---------------------------------------------------------------------------
// Fragment helpers (contiguous k order, k = 16*hf + i, wave32)
// ---------------------------------------------------------------------------

// A fragment 16x32 f16 from f32 row-major. rowp points at A[row] + k0.
// lane<16: elems 0..15 = K k0..k0+15 ; lanes 16..31 get +16.
__device__ inline v16h a_frag_f32(const float* __restrict__ rowp, int hf) {
  v16h a;
  const float* p = rowp + 16 * hf;
#pragma unroll
  for (int i = 0; i < 8; ++i) a[i] = (_Float16)p[i];
#pragma unroll
  for (int i = 0; i < 8; ++i) a[8 + i] = (_Float16)p[8 + i];
  return a;
}

// B fragment 32x16 f16 from f32 "B^T row": colp points at BT[n] + k0 + 16*hf.
__device__ inline v16h b_frag_f32(const float* __restrict__ colp) {
  v16h b;
#pragma unroll
  for (int i = 0; i < 16; ++i) b[i] = (_Float16)colp[i];
  return b;
}

__device__ inline float siluf(float x)     { return x / (1.0f + __expf(-x)); }
__device__ inline float softplusf(float x) { return x > 20.0f ? x : log1pf(__expf(x)); }

// ---------------------------------------------------------------------------
// Weight convert + transpose:  W[k][n] (f32) -> Wt[n][k] (f16)
// ---------------------------------------------------------------------------
__global__ void k_wt_convert(const float* __restrict__ W, _Float16* __restrict__ Wt,
                             int K, int N) {
  int idx = blockIdx.x * 256 + threadIdx.x;
  if (idx >= K * N) return;
  int n = idx / K, k = idx % K;
  Wt[idx] = (_Float16)W[(size_t)k * N + n];
}

// ---------------------------------------------------------------------------
// WMMA GEMM: C[M,N] = A[M,K](f32, lda) * Wt[N,K](f16)^T + bias
// grid = (N/64, M/128), block = 128 (4 waves).
// Each wave computes a 32(M) x 64(N) tile: 2 A-fragments reuse 4 B-fragments
// -> 8 WMMAs per k-step, ~16 FLOP/byte of loaded operands.
// ---------------------------------------------------------------------------
__global__ __launch_bounds__(128) void k_gemm(const float* __restrict__ A, int lda,
                                              const _Float16* __restrict__ Wt,
                                              const float* __restrict__ bias,
                                              float* __restrict__ C, int ldc, int K) {
  const int lane = threadIdx.x & 31;
  const int wave = threadIdx.x >> 5;
  const int lm = lane & 15, hf = lane >> 4;
  const int nb0 = blockIdx.x * 64;
  const int mb  = blockIdx.y * 128 + wave * 32;
  v8f acc0[4] = {};
  v8f acc1[4] = {};
  const float* arow0 = A + (size_t)(mb + lm) * lda;
  const float* arow1 = A + (size_t)(mb + 16 + lm) * lda;
  for (int k0 = 0; k0 < K; k0 += 32) {
    v16h a0 = a_frag_f32(arow0 + k0, hf);
    v16h a1 = a_frag_f32(arow1 + k0, hf);
#pragma unroll
    for (int j = 0; j < 4; ++j) {
      const _Float16* bp = Wt + (size_t)(nb0 + 16 * j + lm) * K + k0 + 16 * hf;
      v16h b = *(const v16h*)bp;                 // 32B aligned: K,k0 multiples of 32/16
      acc0[j] = WMMA_F32_F16(a0, b, acc0[j]);
      acc1[j] = WMMA_F32_F16(a1, b, acc1[j]);
    }
  }
#pragma unroll
  for (int j = 0; j < 4; ++j) {
    const int col = nb0 + 16 * j + lm;
    const float bv = bias ? bias[col] : 0.0f;
    const int r0 = mb + 8 * hf;
    const int r1 = mb + 16 + 8 * hf;
#pragma unroll
    for (int r = 0; r < 8; ++r) {
      C[(size_t)(r0 + r) * ldc + col] = acc0[j][r] + bv;
      C[(size_t)(r1 + r) * ldc + col] = acc1[j][r] + bv;
    }
  }
}

// ---------------------------------------------------------------------------
// Embedding: token conv(3-tap, wrap pad) + sinusoidal pos + mark proj
// ---------------------------------------------------------------------------
__global__ void k_embed(const float* __restrict__ xd, const float* __restrict__ xm,
                        const float* __restrict__ tokw, const float* __restrict__ tempw,
                        float* __restrict__ X) {
  int idx = blockIdx.x * 256 + threadIdx.x;
  if (idx >= MROWS * DMODEL) return;
  int d = idx % DMODEL;
  int bl = idx / DMODEL;
  int l = bl % LSEQ, b = bl / LSEQ;
  float acc = 0.0f;
#pragma unroll
  for (int k = 0; k < 3; ++k) {
    int src = l + k - 1;
    if (src < 0) src = LSEQ - 1;
    if (src >= LSEQ) src = 0;
    const float* xr = xd + ((size_t)b * LSEQ + src) * 7;
    const float* tw = tokw + (size_t)k * 7 * DMODEL + d;
#pragma unroll
    for (int c = 0; c < 7; ++c) acc += xr[c] * tw[(size_t)c * DMODEL];
  }
  const float* xmr = xm + ((size_t)b * LSEQ + l) * 4;
#pragma unroll
  for (int c = 0; c < 4; ++c) acc += xmr[c] * tempw[c * DMODEL + d];
  int i2 = d & ~1;
  float div = __expf(-(float)i2 * (9.210340371976184f / (float)DMODEL)); // ln(1e4)
  float ang = (float)l * div;
  acc += (d & 1) ? cosf(ang) : sinf(ang);
  X[idx] = acc;
}

// ---------------------------------------------------------------------------
// Depthwise causal conv (4 taps) + SiLU on u-half of XZ
// ---------------------------------------------------------------------------
__global__ void k_conv_silu(const float* __restrict__ XZ, const float* __restrict__ cw,
                            const float* __restrict__ cb, float* __restrict__ UC) {
  int idx = blockIdx.x * 256 + threadIdx.x;
  if (idx >= MROWS * DINNER) return;
  int d = idx % DINNER;
  int bl = idx / DINNER;
  int l = bl % LSEQ, b = bl / LSEQ;
  const float* ub = XZ + (size_t)b * LSEQ * (2 * DINNER) + d;
  float acc = cb[d];
#pragma unroll
  for (int k = 0; k < 4; ++k) {
    int ls = l + k - 3;
    if (ls >= 0) acc += ub[(size_t)ls * (2 * DINNER)] * cw[d * 4 + k];
  }
  UC[idx] = siluf(acc);
}

__global__ void k_softplus(float* __restrict__ D, int n) {
  int idx = blockIdx.x * 256 + threadIdx.x;
  if (idx < n) D[idx] = softplusf(D[idx]);
}

// ---------------------------------------------------------------------------
// Selective scan: one lane per (b, d) channel; 16 states in registers.
// Writes y in-place over delta.
// ---------------------------------------------------------------------------
__global__ __launch_bounds__(256) void k_scan(const float* __restrict__ delta,
                                              const float* __restrict__ uc,
                                              const float* __restrict__ dbc,
                                              const float* __restrict__ A_log,
                                              float* __restrict__ Y) {
  int idx = blockIdx.x * 256 + threadIdx.x;   // BBATCH*DINNER = 8192
  if (idx >= BBATCH * DINNER) return;
  int d = idx % DINNER, b = idx / DINNER;
  float A[DSTATE], h[DSTATE];
#pragma unroll
  for (int s = 0; s < DSTATE; ++s) { A[s] = -__expf(A_log[d * DSTATE + s]); h[s] = 0.0f; }
  const float* drow = delta + (size_t)b * LSEQ * DINNER + d;
  const float* urow = uc    + (size_t)b * LSEQ * DINNER + d;
  const float* brow = dbc   + (size_t)b * LSEQ * 64;
  float* yrow = Y + (size_t)b * LSEQ * DINNER + d;
  for (int l = 0; l < LSEQ; ++l) {
    float dt = drow[(size_t)l * DINNER];
    float du = dt * urow[(size_t)l * DINNER];
    const float* bc = brow + (size_t)l * 64;
    float y = 0.0f;
#pragma unroll
    for (int s = 0; s < DSTATE; ++s) {
      h[s] = __expf(dt * A[s]) * h[s] + du * bc[DTRANK + s];
      y += h[s] * bc[DTRANK + DSTATE + s];
    }
    yrow[(size_t)l * DINNER] = y;
  }
}

// ---------------------------------------------------------------------------
// Gate: UC := (Y + UC*D_skip) * silu(z)   (z = second half of XZ)
// ---------------------------------------------------------------------------
__global__ void k_gate(const float* __restrict__ Y, const float* __restrict__ XZ,
                       const float* __restrict__ Dskip, float* __restrict__ UC) {
  int idx = blockIdx.x * 256 + threadIdx.x;
  if (idx >= MROWS * DINNER) return;
  int d = idx % DINNER;
  size_t row = idx / DINNER;
  float z = XZ[row * (2 * DINNER) + DINNER + d];
  float y = Y[idx] + UC[idx] * Dskip[d];
  UC[idx] = y * siluf(z);
}

// ---------------------------------------------------------------------------
// Residual add + LayerNorm, in-place on X. One wave per row of 512.
// ---------------------------------------------------------------------------
__global__ __launch_bounds__(32) void k_ln_add(const float* __restrict__ add,
                                               const float* __restrict__ g,
                                               const float* __restrict__ bta,
                                               float* __restrict__ X) {
  int row = blockIdx.x;
  int lane = threadIdx.x;
  float* xr = X + (size_t)row * DMODEL;
  const float* ar = add + (size_t)row * DMODEL;
  float v[16];
  float s = 0.0f;
#pragma unroll
  for (int i = 0; i < 16; ++i) { v[i] = xr[lane + 32 * i] + ar[lane + 32 * i]; s += v[i]; }
#pragma unroll
  for (int m = 16; m >= 1; m >>= 1) s += __shfl_xor(s, m, 32);
  float mu = s * (1.0f / DMODEL);
  float var = 0.0f;
#pragma unroll
  for (int i = 0; i < 16; ++i) { float t = v[i] - mu; var += t * t; }
#pragma unroll
  for (int m = 16; m >= 1; m >>= 1) var += __shfl_xor(var, m, 32);
  float rinv = rsqrtf(var * (1.0f / DMODEL) + 1e-5f);
#pragma unroll
  for (int i = 0; i < 16; ++i) {
    int c = lane + 32 * i;
    xr[c] = (v[i] - mu) * rinv * g[c] + bta[c];
  }
}

// ---------------------------------------------------------------------------
// V transpose per (b,h): VT[b][h][d][l] = V[b,l,h*64+d]
// ---------------------------------------------------------------------------
__global__ void k_transpose_v(const float* __restrict__ V, float* __restrict__ VT) {
  int idx = blockIdx.x * 256 + threadIdx.x;
  if (idx >= BBATCH * NHEADS * DHEAD * LSEQ) return;
  int l = idx % LSEQ;
  int d = (idx / LSEQ) % DHEAD;
  int h = (idx / (LSEQ * DHEAD)) % NHEADS;
  int b = idx / (LSEQ * DHEAD * NHEADS);
  VT[idx] = V[((size_t)b * LSEQ + l) * DMODEL + h * DHEAD + d];
}

// ---------------------------------------------------------------------------
// Flash attention (causal): one wave per (query-tile of 16, head, batch).
// Key blocks of 32; QK^T and P*V both via v_wmma_f32_16x16x32_f16.
// ---------------------------------------------------------------------------
__global__ __launch_bounds__(32) void k_flash(const float* __restrict__ Q,
                                              const float* __restrict__ Km,
                                              const float* __restrict__ VT,
                                              float* __restrict__ O) {
  __shared__ float    sS[16][32];
  __shared__ _Float16 sP[16][32];
  __shared__ float    sAlpha[16];
  __shared__ float    sStat[16];
  const int lane = threadIdx.x;
  const int lm = lane & 15, hf = lane >> 4;
  const int qt = blockIdx.x, h = blockIdx.y, b = blockIdx.z;

  // Q fragments, pre-scaled by 1/sqrt(dh)=0.125
  v16h qa[2];
  const float* qrow = Q + ((size_t)b * LSEQ + qt * 16 + lm) * DMODEL + h * DHEAD;
#pragma unroll
  for (int kk = 0; kk < 2; ++kk) {
    const float* p = qrow + kk * 32 + 16 * hf;
#pragma unroll
    for (int i = 0; i < 8; ++i) qa[kk][i]     = (_Float16)(p[i] * 0.125f);
#pragma unroll
    for (int i = 0; i < 8; ++i) qa[kk][8 + i] = (_Float16)(p[8 + i] * 0.125f);
  }

  v8f oacc[4] = {};
  float m_i = -1e30f, l_i = 0.0f;           // meaningful in lanes 0..15
  const int qlast = qt * 16 + 15;

  for (int s0 = 0; s0 <= qlast; s0 += 32) {
    // S = Q K^T over dh=64 (two K=32 steps), two 16-col halves
    v8f sacc[2] = {};
#pragma unroll
    for (int kk = 0; kk < 2; ++kk) {
#pragma unroll
      for (int j = 0; j < 2; ++j) {
        const float* kp = Km + ((size_t)b * LSEQ + s0 + 16 * j + lm) * DMODEL +
                          h * DHEAD + kk * 32 + 16 * hf;
        v16h kb = b_frag_f32(kp);
        sacc[j] = WMMA_F32_F16(qa[kk], kb, sacc[j]);
      }
    }
#pragma unroll
    for (int j = 0; j < 2; ++j)
#pragma unroll
      for (int r = 0; r < 8; ++r) sS[r + 8 * hf][16 * j + lm] = sacc[j][r];
    __syncthreads();

    // Online softmax: lanes 0..15 each own one query row
    if (lane < 16) {
      int mg = qt * 16 + lane;
      float sv[32];
      float rmax = -1e30f;
#pragma unroll
      for (int c = 0; c < 32; ++c) {
        float x = sS[lane][c];
        if (s0 + c > mg) x = -1e30f;        // strict causal mask
        sv[c] = x;
        rmax = fmaxf(rmax, x);
      }
      float mnew = fmaxf(m_i, rmax);
      float rsum = 0.0f;
#pragma unroll
      for (int c = 0; c < 32; ++c) {
        float p = __expf(sv[c] - mnew);
        sP[lane][c] = (_Float16)p;
        rsum += p;
      }
      float alpha = __expf(m_i - mnew);
      l_i = l_i * alpha + rsum;
      m_i = mnew;
      sAlpha[lane] = alpha;
    }
    __syncthreads();

    // Rescale O accumulators by alpha[row]
    float al[8];
#pragma unroll
    for (int r = 0; r < 8; ++r) al[r] = sAlpha[r + 8 * hf];
#pragma unroll
    for (int j = 0; j < 4; ++j)
#pragma unroll
      for (int r = 0; r < 8; ++r) oacc[j][r] *= al[r];

    // P as A-fragment (16x32) from LDS
    v16h pa;
    {
      const _Float16* pp = &sP[lm][16 * hf];
#pragma unroll
      for (int i = 0; i < 8; ++i) { pa[i] = pp[i]; pa[8 + i] = pp[8 + i]; }
    }
    // O += P * V   (V^T rows are contiguous in s)
#pragma unroll
    for (int j = 0; j < 4; ++j) {
      const float* vp = VT + ((size_t)(b * NHEADS + h) * DHEAD + 16 * j + lm) * LSEQ +
                        s0 + 16 * hf;
      v16h vb = b_frag_f32(vp);
      oacc[j] = WMMA_F32_F16(pa, vb, oacc[j]);
    }
    __syncthreads();
  }

  if (lane < 16) sStat[lane] = 1.0f / l_i;
  __syncthreads();
  float li[8];
#pragma unroll
  for (int r = 0; r < 8; ++r) li[r] = sStat[r + 8 * hf];
#pragma unroll
  for (int j = 0; j < 4; ++j)
#pragma unroll
    for (int r = 0; r < 8; ++r)
      O[((size_t)b * LSEQ + qt * 16 + r + 8 * hf) * DMODEL + h * DHEAD + 16 * j + lm] =
          oacc[j][r] * li[r];
}

// ---------------------------------------------------------------------------
// Final projection: out[row, c] = X[row] . proj_w[:, c] + proj_b[c], c < 7
// ---------------------------------------------------------------------------
__global__ void k_final_proj(const float* __restrict__ X, const float* __restrict__ pw,
                             const float* __restrict__ pb, float* __restrict__ out) {
  int idx = blockIdx.x * 256 + threadIdx.x;
  if (idx >= MROWS * 7) return;
  int c = idx % 7;
  size_t row = idx / 7;
  const float* xr = X + row * DMODEL;
  float acc = pb[c];
  for (int d = 0; d < DMODEL; ++d) acc += xr[d] * pw[d * 7 + c];
  out[idx] = acc;
}

// ---------------------------------------------------------------------------
// Host orchestration
// ---------------------------------------------------------------------------
extern "C" void kernel_launch(void* const* d_in, const int* in_sizes, int n_in,
                              void* d_out, int out_size, void* d_ws, size_t ws_size,
                              hipStream_t stream) {
  (void)in_sizes; (void)n_in; (void)out_size; (void)ws_size;
  const float* x_dec    = (const float*)d_in[0];
  const float* x_mark   = (const float*)d_in[1];
  const float* token_w  = (const float*)d_in[2];
  const float* temp_w   = (const float*)d_in[3];
  const float* in_proj_w= (const float*)d_in[4];
  const float* conv_w   = (const float*)d_in[5];
  const float* conv_b   = (const float*)d_in[6];
  const float* x_proj_w = (const float*)d_in[7];
  const float* dt_w     = (const float*)d_in[8];
  const float* dt_b     = (const float*)d_in[9];
  const float* A_log    = (const float*)d_in[10];
  const float* D_skip   = (const float*)d_in[11];
  const float* out_w_m  = (const float*)d_in[12];
  const float* wq = (const float*)d_in[13]; const float* bq = (const float*)d_in[14];
  const float* wk = (const float*)d_in[15]; const float* bk = (const float*)d_in[16];
  const float* wv = (const float*)d_in[17]; const float* bv = (const float*)d_in[18];
  const float* wo = (const float*)d_in[19]; const float* bo = (const float*)d_in[20];
  const float* ln1_g = (const float*)d_in[21]; const float* ln1_b = (const float*)d_in[22];
  const float* ln2_g = (const float*)d_in[23]; const float* ln2_b = (const float*)d_in[24];
  const float* proj_w = (const float*)d_in[25]; const float* proj_b = (const float*)d_in[26];

  const size_t MB = 1ull << 20;
  char* ws = (char*)d_ws;
  float*     X     = (float*)(ws + 0);          // 16 MB
  float*     XZ    = (float*)(ws + 16 * MB);    // 64 MB (xz; later Q/K/V/VT/O)
  float*     UC    = (float*)(ws + 80 * MB);    // 32 MB (uc; later gated y)
  float*     DBC   = (float*)(ws + 112 * MB);   //  2 MB
  float*     DELTA = (float*)(ws + 114 * MB);   // 32 MB (delta -> y in-place; later GEMM-out)
  _Float16*  WT    = (_Float16*)(ws + 146 * MB);//  2 MB f16 transposed weights
  float* GOUT = DELTA;
  float* Qb = XZ;
  float* Kb = (float*)(ws + 32 * MB);
  float* Vb = (float*)(ws + 48 * MB);
  float* VTb = (float*)(ws + 64 * MB);
  float* Ob = Vb;   // V region reusable after transpose

  auto blks = [](size_t n) { return dim3((unsigned)((n + 255) / 256)); };

  // Embedding
  k_embed<<<blks((size_t)MROWS * DMODEL), 256, 0, stream>>>(x_dec, x_mark, token_w, temp_w, X);

  for (int i = 0; i < 2; ++i) {
    const float* ipw = in_proj_w + (size_t)i * DMODEL * 2 * DINNER;
    const float* cwi = conv_w + (size_t)i * DINNER * 4;
    const float* cbi = conv_b + (size_t)i * DINNER;
    const float* xpw = x_proj_w + (size_t)i * DINNER * 64;
    const float* dtwi = dt_w + (size_t)i * DTRANK * DINNER;
    const float* dtbi = dt_b + (size_t)i * DINNER;
    const float* Ali = A_log + (size_t)i * DINNER * DSTATE;
    const float* Dsi = D_skip + (size_t)i * DINNER;
    const float* owi = out_w_m + (size_t)i * DINNER * DMODEL;
    const float* wqi = wq + (size_t)i * DMODEL * DMODEL; const float* bqi = bq + (size_t)i * DMODEL;
    const float* wki = wk + (size_t)i * DMODEL * DMODEL; const float* bki = bk + (size_t)i * DMODEL;
    const float* wvi = wv + (size_t)i * DMODEL * DMODEL; const float* bvi = bv + (size_t)i * DMODEL;
    const float* woi = wo + (size_t)i * DMODEL * DMODEL; const float* boi = bo + (size_t)i * DMODEL;

    // xz = x @ in_proj   (M=8192, N=2048, K=512)
    k_wt_convert<<<blks((size_t)DMODEL * 2 * DINNER), 256, 0, stream>>>(ipw, WT, DMODEL, 2 * DINNER);
    k_gemm<<<dim3(2 * DINNER / 64, MROWS / 128), 128, 0, stream>>>(X, DMODEL, WT, nullptr, XZ, 2 * DINNER, DMODEL);
    // uc = silu(conv(u))
    k_conv_silu<<<blks((size_t)MROWS * DINNER), 256, 0, stream>>>(XZ, cwi, cbi, UC);
    // dbc = uc @ x_proj  (N=64, K=1024)
    k_wt_convert<<<blks((size_t)DINNER * 64), 256, 0, stream>>>(xpw, WT, DINNER, 64);
    k_gemm<<<dim3(1, MROWS / 128), 128, 0, stream>>>(UC, DINNER, WT, nullptr, DBC, 64, DINNER);
    // delta = softplus(dt @ dt_w + dt_b)  (A = dbc[..., :32], lda=64, K=32)
    k_wt_convert<<<blks((size_t)DTRANK * DINNER), 256, 0, stream>>>(dtwi, WT, DTRANK, DINNER);
    k_gemm<<<dim3(DINNER / 64, MROWS / 128), 128, 0, stream>>>(DBC, 64, WT, dtbi, DELTA, DINNER, DTRANK);
    k_softplus<<<blks((size_t)MROWS * DINNER), 256, 0, stream>>>(DELTA, MROWS * DINNER);
    // selective scan (y overwrites delta)
    k_scan<<<blks((size_t)BBATCH * DINNER), 256, 0, stream>>>(DELTA, UC, DBC, Ali, DELTA);
    // gated y (into UC)
    k_gate<<<blks((size_t)MROWS * DINNER), 256, 0, stream>>>(DELTA, XZ, Dsi, UC);
    // x = LN(x + y @ out_w)
    k_wt_convert<<<blks((size_t)DINNER * DMODEL), 256, 0, stream>>>(owi, WT, DINNER, DMODEL);
    k_gemm<<<dim3(DMODEL / 64, MROWS / 128), 128, 0, stream>>>(UC, DINNER, WT, nullptr, GOUT, DMODEL, DINNER);
    k_ln_add<<<MROWS, 32, 0, stream>>>(GOUT, ln1_g + (size_t)i * DMODEL, ln1_b + (size_t)i * DMODEL, X);

    // q, k, v projections
    k_wt_convert<<<blks((size_t)DMODEL * DMODEL), 256, 0, stream>>>(wqi, WT, DMODEL, DMODEL);
    k_gemm<<<dim3(DMODEL / 64, MROWS / 128), 128, 0, stream>>>(X, DMODEL, WT, bqi, Qb, DMODEL, DMODEL);
    k_wt_convert<<<blks((size_t)DMODEL * DMODEL), 256, 0, stream>>>(wki, WT, DMODEL, DMODEL);
    k_gemm<<<dim3(DMODEL / 64, MROWS / 128), 128, 0, stream>>>(X, DMODEL, WT, bki, Kb, DMODEL, DMODEL);
    k_wt_convert<<<blks((size_t)DMODEL * DMODEL), 256, 0, stream>>>(wvi, WT, DMODEL, DMODEL);
    k_gemm<<<dim3(DMODEL / 64, MROWS / 128), 128, 0, stream>>>(X, DMODEL, WT, bvi, Vb, DMODEL, DMODEL);
    // V -> V^T per (b,h)
    k_transpose_v<<<blks((size_t)BBATCH * NHEADS * DHEAD * LSEQ), 256, 0, stream>>>(Vb, VTb);
    // flash attention (O reuses V region)
    k_flash<<<dim3(LSEQ / 16, NHEADS, BBATCH), 32, 0, stream>>>(Qb, Kb, VTb, Ob);
    // x = LN(x + o @ wo + bo)
    k_wt_convert<<<blks((size_t)DMODEL * DMODEL), 256, 0, stream>>>(woi, WT, DMODEL, DMODEL);
    k_gemm<<<dim3(DMODEL / 64, MROWS / 128), 128, 0, stream>>>(Ob, DMODEL, WT, boi, GOUT, DMODEL, DMODEL);
    k_ln_add<<<MROWS, 32, 0, stream>>>(GOUT, ln2_g + (size_t)i * DMODEL, ln2_b + (size_t)i * DMODEL, X);
  }

  // Final projection to C_OUT=7
  k_final_proj<<<blks((size_t)MROWS * 7), 256, 0, stream>>>(X, proj_w, proj_b, (float*)d_out);
}
